// AttentionLayer_43619687858179
// MI455X (gfx1250) — matrix operand order repaired and checked by reading, not compiled, not run on hardware
//
#include <hip/hip_runtime.h>
#include <math.h>
#include <stdint.h>

#define B_ 16
#define S_ 2048
#define H_ 256
#define LOG2E 1.4426950408889634f
#define QPITCH 264   // Q rows in LDS: 256 + 8-half pad
#define KPITCH 264   // K tile rows in LDS
#define VPITCH 40    // V^T tile rows in LDS: 32 + 8-half pad

typedef __attribute__((ext_vector_type(16))) _Float16 v16h;
typedef __attribute__((ext_vector_type(8)))  _Float16 h8;
typedef __attribute__((ext_vector_type(8)))  float    v8f;

union U16h { v16h v; h8 h[2]; };

// ---------------- projection kernels (negligible cost) ----------------
__global__ void proj_qk_kernel(const float* __restrict__ forces,
                               const float* __restrict__ noisy,
                               const float* __restrict__ Wq, const float* __restrict__ bq,
                               const float* __restrict__ Wk, const float* __restrict__ bk,
                               _Float16* __restrict__ Qh, _Float16* __restrict__ Kh) {
  const int h  = threadIdx.x;      // blockDim.x == H_
  const int sb = blockIdx.x;       // b*S + s
  const float* f = forces + (size_t)sb * 3;
  const float* n = noisy  + (size_t)sb * 3;
  float q = fmaf(f[0], Wq[h], fmaf(f[1], Wq[H_ + h], fmaf(f[2], Wq[2 * H_ + h], bq[h])));
  float k = fmaf(n[0], Wk[h], fmaf(n[1], Wk[H_ + h], fmaf(n[2], Wk[2 * H_ + h], bk[h])));
  const size_t o = (size_t)sb * H_ + h;
  Qh[o] = (_Float16)q;
  Kh[o] = (_Float16)k;
}

__global__ void proj_vt_kernel(const float* __restrict__ noisy,
                               const float* __restrict__ Wv, const float* __restrict__ bv,
                               _Float16* __restrict__ Vt) {
  const int t  = threadIdx.x;              // 256
  const int sc = blockIdx.x & 7;           // S_/256 chunks
  const int h  = (blockIdx.x >> 3) & 255;
  const int b  = blockIdx.x >> 11;
  const int s  = sc * 256 + t;
  const float* n = noisy + ((size_t)b * S_ + s) * 3;
  float v = fmaf(n[0], Wv[h], fmaf(n[1], Wv[H_ + h], fmaf(n[2], Wv[2 * H_ + h], bv[h])));
  Vt[((size_t)b * H_ + h) * S_ + s] = (_Float16)v;   // coalesced along s
}

// ---------------- WMMA operand loaders ----------------
// A operand (16-bit 16x32): lane row M=lane&15; 16 contiguous K halves at k0+16*hi
__device__ __forceinline__ v16h load_a(const _Float16* row, int k0, int hi8) {
  U16h u;
  u.h[0] = *(const h8*)(row + k0 + 2 * hi8);
  u.h[1] = *(const h8*)(row + k0 + 2 * hi8 + 8);
  return u.v;
}
// B operand (16-bit 32x16): lane col N=lane&15; 16 contiguous K halves at k0+16*hi
__device__ __forceinline__ v16h load_b(const _Float16* row, int k0, int hi16) {
  U16h u;
  const _Float16* p = row + k0 + hi16;
  u.h[0] = *(const h8*)(p);
  u.h[1] = *(const h8*)(p + 8);
  return u.v;
}

// ---------------- async global->LDS copy (CDNA5, tracked by ASYNCcnt) ----------------
__device__ __forceinline__ void async_copy16(uint32_t lds_byte, const void* gptr) {
  asm volatile("global_load_async_to_lds_b128 %0, %1, off"
               :: "v"(lds_byte), "v"((uint64_t)(uintptr_t)gptr)
               : "memory");
}
__device__ __forceinline__ void wait_async0() {
  asm volatile("s_wait_asynccnt 0x0" ::: "memory");
}

// Issue one kv tile (K: 32x256 halves, V^T: 256x32 halves) into LDS buffers.
// 128 threads x 16B x 8 iters per matrix = 16KB each.
__device__ __forceinline__ void issue_tile(const _Float16* Kb, const _Float16* Vb, int kv0,
                                           uint32_t ksLds, uint32_t vtLds, int tid) {
#pragma unroll
  for (int i = 0; i < 8; ++i) {
    int f = i * 128 + tid;            // 16B unit
    int row = f >> 5, col = f & 31;   // K tile: 32 rows x 32 cols of 16B
    async_copy16(ksLds + (uint32_t)(row * (KPITCH * 2) + col * 16),
                 (const char*)(Kb + (size_t)(kv0 + row) * H_) + col * 16);
  }
#pragma unroll
  for (int i = 0; i < 8; ++i) {
    int f = i * 128 + tid;
    int row = f >> 2, col = f & 3;    // V^T tile: 256 rows x 4 cols of 16B
    async_copy16(vtLds + (uint32_t)(row * (VPITCH * 2) + col * 16),
                 (const char*)(Vb + (size_t)row * S_ + kv0) + col * 16);
  }
}

// ---------------- flash attention (transposed-scores formulation) ----------------
__global__ __launch_bounds__(128)
void flash_attn_kernel(const _Float16* __restrict__ Qh,
                       const _Float16* __restrict__ Kh,
                       const _Float16* __restrict__ Vt,
                       float* __restrict__ out) {
  __shared__ __attribute__((aligned(16))) _Float16 Qs[4 * 16 * QPITCH];
  __shared__ __attribute__((aligned(16))) _Float16 Ks[2][32 * KPITCH];
  __shared__ __attribute__((aligned(16))) _Float16 Vts[2][256 * VPITCH];

  const int tid  = threadIdx.x;
  const int lane = tid & 31;
  const int wv   = tid >> 5;
  const int b    = blockIdx.x >> 5;                       // 32 q-tiles per batch
  const int q0   = ((blockIdx.x & 31) << 6) + (wv << 4);  // this wave's 16 q rows
  const int lq   = lane & 15;
  const int hiG  = lane >> 4;
  const int hi8  = hiG << 3;
  const int hi16 = hiG << 4;

  const _Float16* Kb = Kh + (size_t)b * S_ * H_;
  const _Float16* Vb = Vt + (size_t)b * H_ * S_;

  const uint32_t ksLds[2] = { (uint32_t)(uintptr_t)&Ks[0][0],  (uint32_t)(uintptr_t)&Ks[1][0] };
  const uint32_t vtLds[2] = { (uint32_t)(uintptr_t)&Vts[0][0], (uint32_t)(uintptr_t)&Vts[1][0] };

  // prefetch first kv tile while we stage Q
  issue_tile(Kb, Vb, 0, ksLds[0], vtLds[0], tid);

  // stage this wave's Q rows (f16, row-major) into LDS
  const _Float16* Qg = Qh + ((size_t)b * S_ + q0) * H_;
  _Float16* qs = &Qs[wv * 16 * QPITCH];
  for (int i = 0; i < 16; ++i)
    *(h8*)(qs + i * QPITCH + lane * 8) = *(const h8*)(Qg + i * H_ + lane * 8);

  const _Float16* qrow = qs + lq * QPITCH;          // B-operand source: row q = lane&15

  v8f ct[16];                                       // ctx^T tiles: lane=q, VGPR row=h
  v8f z8 = {0.f, 0.f, 0.f, 0.f, 0.f, 0.f, 0.f, 0.f};
#pragma unroll
  for (int t = 0; t < 16; ++t) ct[t] = z8;
  float m_run = -__builtin_inff();
  float l_run = 0.f;

  for (int it = 0; it < S_ / 32; ++it) {
    const int kv0 = it * 32;
    const int cur = it & 1;

    wait_async0();         // my async copies for tile `it` (and Q stage via dscnt below) done
    __syncthreads();       // everyone's copies done; prev tile fully consumed by all waves

    if (kv0 + 32 < S_)     // uniform branch: prefetch next tile into the other buffer
      issue_tile(Kb, Vb, kv0 + 32, ksLds[cur ^ 1], vtLds[cur ^ 1], tid);

    const _Float16* ks = &Ks[cur][0];
    const _Float16* vs = &Vts[cur][0];
    const _Float16* k0r = ks + lq * KPITCH;         // kv rows kv0+lq
    const _Float16* k1r = ks + (16 + lq) * KPITCH;  // kv rows kv0+16+lq

    // scores^T tiles: s0 = kv rows 0..15, s1 = 16..31 of this tile ; cols = q
    v8f s0 = z8, s1 = z8;
#pragma unroll
    for (int c = 0; c < 8; ++c) {                   // K-dim = H = 256 -> 8 chunks of 32
      v16h qb = load_b(qrow, 32 * c, hi16);
      v16h ka = load_a(k0r, 32 * c, hi8);
      s0 = __builtin_amdgcn_wmma_f32_16x16x32_f16(false, ka, false, qb, (short)0, s0, false, false);
      v16h kc = load_a(k1r, 32 * c, hi8);
      s1 = __builtin_amdgcn_wmma_f32_16x16x32_f16(false, kc, false, qb, (short)0, s1, false, false);
    }

    // online softmax over kv: per-lane 16 values + one lane-pair exchange
    float tm = -__builtin_inff();
#pragma unroll
    for (int i = 0; i < 8; ++i) { tm = fmaxf(tm, s0[i]); tm = fmaxf(tm, s1[i]); }
    tm = fmaxf(tm, __shfl_xor(tm, 16, 32));
    const float m_new = fmaxf(m_run, tm);
    const float scale = exp2f((m_run - m_new) * LOG2E);

    float p0[8], p1[8];
    float ts = 0.f;
#pragma unroll
    for (int i = 0; i < 8; ++i) {
      p0[i] = exp2f((s0[i] - m_new) * LOG2E);
      p1[i] = exp2f((s1[i] - m_new) * LOG2E);
      ts += p0[i] + p1[i];
    }
    ts += __shfl_xor(ts, 16, 32);
    l_run = l_run * scale + ts;
    m_run = m_new;

    // Build P^T B-operand: lane-lo needs kv 0..15 (8..15 from hi partner),
    // lane-hi needs kv 16..31 (16..23 from lo partner).
    U16h pb;
#pragma unroll
    for (int i = 0; i < 8; ++i) {
      float send = hiG ? p0[i] : p1[i];             // what the partner needs
      float oth  = __shfl_xor(send, 16, 32);
      pb.h[0][i] = (_Float16)(hiG ? oth : p0[i]);
      pb.h[1][i] = (_Float16)(hiG ? p1[i] : oth);
    }

    // rescale accumulators, then ctx^T += V^T(chunk) x P^T
#pragma unroll
    for (int t = 0; t < 16; ++t) {
#pragma unroll
      for (int j = 0; j < 8; ++j) ct[t][j] *= scale;
    }
#pragma unroll
    for (int t = 0; t < 16; ++t) {
      const _Float16* vrow = vs + (16 * t + lq) * VPITCH;  // V^T row h = 16t + (lane&15)
      v16h va = load_a(vrow, 0, hi8);
      ct[t] = __builtin_amdgcn_wmma_f32_16x16x32_f16(false, va, false, pb.v, (short)0, ct[t], false, false);
    }
  }

  // epilogue: divide by row sum; each lane owns half the h-values of row q=lane&15
  const float inv = 1.0f / l_run;
  float* orow = out + ((size_t)b * S_ + q0 + lq) * H_;
#pragma unroll
  for (int t = 0; t < 16; ++t) {
#pragma unroll
    for (int j = 0; j < 8; ++j)
      orow[16 * t + hi8 + j] = ct[t][j] * inv;      // consecutive j -> b128 stores
  }
}

// ---------------- launcher ----------------
extern "C" void kernel_launch(void* const* d_in, const int* in_sizes, int n_in,
                              void* d_out, int out_size, void* d_ws, size_t ws_size,
                              hipStream_t stream) {
  const float* forces = (const float*)d_in[0];
  const float* noisy  = (const float*)d_in[1];
  const float* Wq     = (const float*)d_in[2];
  const float* bq     = (const float*)d_in[3];
  const float* Wk     = (const float*)d_in[4];
  const float* bk     = (const float*)d_in[5];
  const float* Wv     = (const float*)d_in[6];
  const float* bv     = (const float*)d_in[7];

  const size_t MAT = (size_t)B_ * S_ * H_;          // elements per staged matrix
  _Float16* Qh = (_Float16*)d_ws;
  _Float16* Kh = Qh + MAT;
  _Float16* Vt = Kh + MAT;                           // needs 3*MAT*2 = 48 MiB of ws

  proj_qk_kernel<<<B_ * S_, H_, 0, stream>>>(forces, noisy, Wq, bq, Wk, bk, Qh, Kh);
  proj_vt_kernel<<<B_ * H_ * (S_ / 256), 256, 0, stream>>>(noisy, Wv, bv, Vt);
  flash_attn_kernel<<<B_ * (S_ / 64), 128, 0, stream>>>(Qh, Kh, Vt, (float*)d_out);
}
